// ACTFromCell_89060441850258
// MI455X (gfx1250) — compile-verified, matrix operand order checked by reading, not run on hardware
//
#include <hip/hip_runtime.h>
#include <hip/hip_bf16.h>
#include <math.h>

// ACT RNN cell (adaptive computation time) for MI455X / gfx1250.
//  T=32, B=128, D=256, H=768, MAX_PONDER=8.
//  1) whh_swizzle_bf16: W_hh fp32 -> bf16 pre-swizzled into WMMA B-fragment
//     order [nt][kt][lane][16] (one contiguous 32B chunk per lane).
//  2) precompute_pre: pre = x @ W_ih[:, :D]^T + b_ih + b_hh (parallel WMMA).
//  3) act_main: 8 persistent workgroups x 16 batch rows; full T x 8 ACT loop
//     in-kernel. hx lives in LDS as bf16 in A-fragment order; the GEMM inner
//     loop is software-pipelined (ping-pong register buffers) so a full
//     prefetch batch is always in flight over the v_wmma work.

#define T_  32
#define B_  128
#define D_  256
#define H_  768
#define MAXP 8
#define EPS_ 0.01f
#define SROW 772           // fp32 LDS row stride (768 + 4 pad)
#define NKT  (H_ / 32)     // 24 k-steps
#define NNT  (H_ / 16)     // 48 n-tiles

typedef __bf16 bf16;
typedef __attribute__((ext_vector_type(16))) __bf16 v16bf;
typedef __attribute__((ext_vector_type(8)))  float  v8f;

union BF16Bits { unsigned short u; bf16 b; };

static __device__ __forceinline__ unsigned short f2bf_bits(float f) {
  union { float f; unsigned u; } a; a.f = f;
  unsigned r = a.u + 0x7FFFu + ((a.u >> 16) & 1u);   // round-to-nearest-even
  return (unsigned short)(r >> 16);
}
static __device__ __forceinline__ bf16 f2bf(float f) {
  BF16Bits c; c.u = f2bf_bits(f); return c.b;
}

// Fragment swizzle: element (row m in 0..15, k in 0..H-1) of a 16xK A/B tile
// lives at linear index ((kt*32 + lane)*16 + j):
//   kt = k/32, r5 = k%32, lane = m + 16*((r5>>3)&1), j = (r5&7) | ((r5>>1)&8)
static __device__ __forceinline__ int frag_idx(int m, int k) {
  const int kt = k >> 5, r5 = k & 31;
  const int lane = m + (((r5 >> 3) & 1) << 4);
  const int j = (r5 & 7) | ((r5 >> 1) & 8);
  return ((kt * 32 + lane) << 4) + j;
}

static __device__ __forceinline__ v16bf ldfrag(const bf16* p) {
  return *reinterpret_cast<const v16bf*>(p);
}

// ---------------------------------------------------------------------------
// W_hh fp32 [H][H] -> bf16 swizzled [nt][kt][lane][16]
// ---------------------------------------------------------------------------
__global__ void whh_swizzle_bf16(const float* __restrict__ w,
                                 unsigned short* __restrict__ o, int n) {
  int i = blockIdx.x * blockDim.x + threadIdx.x;
  if (i >= n) return;
  const int j    = i & 15;
  const int lane = (i >> 4) & 31;
  const int blk  = i >> 9;           // nt*NKT + kt
  const int kt   = blk % NKT;
  const int nt   = blk / NKT;
  const int row  = nt * 16 + (lane & 15);                  // W_hh row (= N col)
  const int k    = kt * 32 + (((lane >> 4) & 1) << 3) + ((j < 8) ? j : j + 8);
  o[i] = f2bf_bits(w[(size_t)row * H_ + k]);
}

// ---------------------------------------------------------------------------
// pre[r, n] = sum_d x[r, d] * W_ih[n, d] + b_ih[n] + b_hh[n]
// One 16x16 output tile per wave, K = 256 (on-the-fly bf16 conversion).
// ---------------------------------------------------------------------------
__global__ void __launch_bounds__(128)
precompute_pre(const float* __restrict__ x,    // [T*B, D]
               const float* __restrict__ Wih,  // [H, D+1]
               const float* __restrict__ bih,
               const float* __restrict__ bhh,
               float* __restrict__ pre)        // [T*B, H]
{
  const int wave = threadIdx.x >> 5;
  const int lane = threadIdx.x & 31;
  const int tile = blockIdx.x * 4 + wave;          // < 256*48
  const int mt = tile / NNT;
  const int nt = tile % NNT;
  const int mrow = mt * 16 + (lane & 15);
  const int ncol = nt * 16 + (lane & 15);
  const int khalf = (lane >= 16) ? 8 : 0;
  const float* xrow = x + (size_t)mrow * D_;
  const float* wrow = Wih + (size_t)ncol * (D_ + 1);

  v8f acc = {};
#pragma unroll 1
  for (int kt = 0; kt < D_ / 32; ++kt) {
    const int kb = kt * 32 + khalf;
    v16bf a, b;
#pragma unroll
    for (int j = 0; j < 16; ++j) {
      const int k = kb + ((j < 8) ? j : j + 8);
      a[j] = f2bf(xrow[k]);
      b[j] = f2bf(wrow[k]);
    }
    acc = __builtin_amdgcn_wmma_f32_16x16x32_bf16(
        false, a, false, b, (short)0, acc, false, false);
  }
  const float bias = bih[ncol] + bhh[ncol];
  const int moff = (lane >= 16) ? 8 : 0;
#pragma unroll
  for (int r = 0; r < 8; ++r) {
    const int m = r + moff;
    pre[(size_t)(mt * 16 + m) * H_ + ncol] = acc[r] + bias;
  }
}

// ---------------------------------------------------------------------------
// Persistent ACT kernel: one workgroup (512 threads = 16 waves) per 16 batch
// rows; runs all T time steps x 8 ponder iterations internally.
// ---------------------------------------------------------------------------
__global__ void __launch_bounds__(512, 1)
act_main(const float* __restrict__ Wih,   // [H, D+1] (flag column)
         const float* __restrict__ wp,    // [H]
         const float* __restrict__ bp,    // [1]
         const bf16*  __restrict__ whh,   // bf16 swizzled [nt][kt][lane][16]
         const float* __restrict__ pre,   // [T*B, H]
         float* __restrict__ out)         // all_hx | hx_last | pcost | steps
{
  extern __shared__ float smem[];
  bf16*  hxbf_s = (bf16*)smem;            // [NKT][32][16] bf16 = 6144 floats
  float* hxn_s  = smem + 6144;            // [16][SROW] hx_new fp32
  float* ahx_s  = hxn_s + 16 * SROW;      // [16][SROW] accum_hx
  float* pre_s  = ahx_s + 16 * SROW;      // [16][SROW] pre[t] tile cache
  float* flag_s = pre_s + 16 * SROW;      // [H] W_ih[:, D]
  float* wp_s   = flag_s + H_;            // [H]
  float* ah_s   = wp_s + H_;              // [16] accum_h
  float* sc_s   = ah_s  + 16;             // [16] step_count
  float* spc_s  = sc_s  + 16;             // [16] step_ponder_cost
  float* p_s    = spc_s + 16;             // [16] p (remainder)
  float* mf_s   = p_s   + 16;             // [16] mask at iter start
  float* pc_s   = mf_s  + 16;             // [16] total ponder cost
  float* mask_s = pc_s  + 16;             // [16] live mask

  const int tid  = threadIdx.x;
  const int lane = tid & 31;
  const int wave = tid >> 5;
  const int gb0  = blockIdx.x * 16;       // first global batch row

  for (int i = tid; i < 6144; i += 512) ((unsigned*)hxbf_s)[i] = 0u;  // hx = 0
  for (int i = tid; i < 16 * SROW; i += 512) ahx_s[i] = 0.f;
  for (int i = tid; i < H_; i += 512) {
    flag_s[i] = Wih[(size_t)i * (D_ + 1) + D_];
    wp_s[i]   = wp[i];
  }
  if (tid < 16) {
    ah_s[tid] = 0.f; sc_s[tid] = 0.f; spc_s[tid] = 0.f;
    pc_s[tid] = 0.f; mask_s[tid] = 1.f;
  }
  __syncthreads();

  const float bp0  = bp[0];
  const int   mloc = lane & 15;
  const int   moff = (lane >= 16) ? 8 : 0;
  const int   nt0  = wave * 3;            // 48 N-tiles / 16 waves

  // loop-invariant fragment base pointers (k-step stride = 512 elements)
  const bf16* aptr = hxbf_s + (lane << 4);
  const bf16* bpt0 = whh + ((size_t)((nt0 + 0) * NKT) * 32 + lane) * 16;
  const bf16* bpt1 = whh + ((size_t)((nt0 + 1) * NKT) * 32 + lane) * 16;
  const bf16* bpt2 = whh + ((size_t)((nt0 + 2) * NKT) * 32 + lane) * 16;

  float* outAll  = out;                                // [T][B][H]
  float* outLast = out + (size_t)T_ * B_ * H_;         // [B][H]
  float* outPc   = outLast + (size_t)B_ * H_;          // [B]
  float* outSt   = outPc + B_;                         // [T][B]

#pragma unroll 1
  for (int t = 0; t < T_; ++t) {
    // cache pre[t] tile in LDS (re-used by all 8 ponder steps)
    {
      const float* pre_t = pre + (size_t)(t * B_ + gb0) * H_;
      for (int i = tid; i < 16 * H_; i += 512)
        pre_s[(i / H_) * SROW + (i % H_)] = pre_t[i];
      if (t + 1 < T_) {   // stage next step's tile into cache (async prefetch)
        const float* pn = pre + (size_t)((t + 1) * B_ + gb0) * H_;
        __builtin_prefetch(pn + tid * 24, 0, 0);
      }
    }
    __syncthreads();

#pragma unroll 1
    for (int ps = 0; ps < MAXP; ++ps) {
      const float flag = (ps == 0) ? 0.f : 1.f;

      // ---- GEMM: hx_new = tanh(hx @ W_hh^T + pre + flag*flagvec) ----------
      // Two-deep software pipeline, ping-pong buffers, 2 k-steps / iter.
      v8f acc0 = {}, acc1 = {}, acc2 = {};
      v16bf aA  = ldfrag(aptr);
      v16bf b0A = ldfrag(bpt0), b1A = ldfrag(bpt1), b2A = ldfrag(bpt2);
#pragma unroll 1
      for (int kt = 0; kt < NKT; kt += 2) {
        const int o1 = (kt + 1) * 512;
        const int o2 = ((kt + 2 < NKT) ? kt + 2 : 0) * 512;  // wrap: harmless
        v16bf aB  = ldfrag(aptr + o1);
        v16bf b0B = ldfrag(bpt0 + o1);
        v16bf b1B = ldfrag(bpt1 + o1);
        v16bf b2B = ldfrag(bpt2 + o1);
        acc0 = __builtin_amdgcn_wmma_f32_16x16x32_bf16(
            false, aA, false, b0A, (short)0, acc0, false, false);
        acc1 = __builtin_amdgcn_wmma_f32_16x16x32_bf16(
            false, aA, false, b1A, (short)0, acc1, false, false);
        acc2 = __builtin_amdgcn_wmma_f32_16x16x32_bf16(
            false, aA, false, b2A, (short)0, acc2, false, false);
        aA  = ldfrag(aptr + o2);
        b0A = ldfrag(bpt0 + o2);
        b1A = ldfrag(bpt1 + o2);
        b2A = ldfrag(bpt2 + o2);
        acc0 = __builtin_amdgcn_wmma_f32_16x16x32_bf16(
            false, aB, false, b0B, (short)0, acc0, false, false);
        acc1 = __builtin_amdgcn_wmma_f32_16x16x32_bf16(
            false, aB, false, b1B, (short)0, acc1, false, false);
        acc2 = __builtin_amdgcn_wmma_f32_16x16x32_bf16(
            false, aB, false, b2B, (short)0, acc2, false, false);
      }
      {
        const v8f accs[3] = {acc0, acc1, acc2};
#pragma unroll
        for (int jt = 0; jt < 3; ++jt) {
          const int col = (nt0 + jt) * 16 + mloc;
          const float fv = flag * flag_s[col];
#pragma unroll
          for (int r = 0; r < 8; ++r) {
            const int m = r + moff;
            const float v = accs[jt][r] + pre_s[m * SROW + col] + fv;
            hxn_s[m * SROW + col] = tanhf(v);
          }
        }
      }
      __syncthreads();

      // ---- halting dot + per-row scalar ACT update (wave w owns row w) ----
      {
        const float* rowp = hxn_s + wave * SROW;
        float part = 0.f;
#pragma unroll
        for (int i = 0; i < H_ / 32; ++i)
          part += rowp[lane + i * 32] * wp_s[lane + i * 32];
#pragma unroll
        for (int off = 16; off > 0; off >>= 1)
          part += __shfl_xor(part, off, 32);
        if (lane == 0) {
          const int r = wave;
          const float mask = mask_s[r];
          const float ah   = ah_s[r];
          if (mask != 0.f) spc_s[r] = -ah;              // pre-update cost
          const float hsig = 1.f / (1.f + expf(-(part + bp0)));
          const float ah2  = ah + mask * hsig;
          const float pp   = hsig - fmaxf(ah2 - 1.f, 0.f);
          sc_s[r]  += mask;
          p_s[r]    = pp;
          mf_s[r]   = mask;
          mask_s[r] = (ah2 < (1.f - EPS_)) ? 1.f : 0.f;
          ah_s[r]   = ah2;
        }
      }
      __syncthreads();

      // ---- ahx += mf*(1+p)*hx_new; hx(bf16, swizzled) = mask ? new : old --
      for (int i = tid; i < 16 * H_; i += 512) {
        const int r = i / H_, c = i % H_;
        const float hn = hxn_s[r * SROW + c];
        const float mf = mf_s[r];
        ahx_s[r * SROW + c] += mf * (1.f + p_s[r]) * hn;
        if (mf != 0.f) hxbf_s[frag_idx(r, c)] = f2bf(hn);
      }
      __syncthreads();
    } // ponder

    // ---- finalize time step: hx_out = ahx / sc ----------------------------
    for (int i = tid; i < 16 * H_; i += 512) {
      const int r = i / H_, c = i % H_;
      const float ho = ahx_s[r * SROW + c] / sc_s[r];
      ahx_s[r * SROW + c] = 0.f;
      hxbf_s[frag_idx(r, c)] = f2bf(ho);               // carry to next t
      outAll[(size_t)(t * B_ + gb0 + r) * H_ + c] = ho;
      if (t == T_ - 1) outLast[(size_t)(gb0 + r) * H_ + c] = ho;
    }
    if (tid < 16) {
      outSt[t * B_ + gb0 + tid] = sc_s[tid];
      pc_s[tid] += spc_s[tid];
      if (t == T_ - 1) outPc[gb0 + tid] = pc_s[tid];
      ah_s[tid] = 0.f; sc_s[tid] = 0.f; spc_s[tid] = 0.f; mask_s[tid] = 1.f;
    }
    __syncthreads();
  } // time
}

// ---------------------------------------------------------------------------
extern "C" void kernel_launch(void* const* d_in, const int* in_sizes, int n_in,
                              void* d_out, int out_size, void* d_ws, size_t ws_size,
                              hipStream_t stream) {
  const float* x   = (const float*)d_in[0];   // [T,B,D]
  const float* Wih = (const float*)d_in[1];   // [H,D+1]
  const float* Whh = (const float*)d_in[2];   // [H,H]
  const float* bih = (const float*)d_in[3];
  const float* bhh = (const float*)d_in[4];
  const float* wp  = (const float*)d_in[5];
  const float* bp  = (const float*)d_in[6];
  float* out = (float*)d_out;

  // workspace: [W_hh bf16 swizzled : 1.18 MB][pre fp32 : 12.6 MB]
  unsigned short* whh_bf = (unsigned short*)d_ws;
  float* pre = (float*)((char*)d_ws + (size_t)H_ * H_ * sizeof(unsigned short));

  whh_swizzle_bf16<<<(H_ * H_ + 255) / 256, 256, 0, stream>>>(
      Whh, whh_bf, H_ * H_);

  const int tiles = (T_ * B_ / 16) * NNT;    // 256 * 48
  precompute_pre<<<tiles / 4, 128, 0, stream>>>(x, Wih, bih, bhh, pre);

  const size_t smem =
      (size_t)(6144 + 3 * 16 * SROW + 2 * H_ + 7 * 16) * sizeof(float);
  act_main<<<B_ / 16, 512, smem, stream>>>(
      Wih, wp, bp, (const bf16*)whh_bf, pre, out);
  (void)in_sizes; (void)n_in; (void)out_size; (void)ws_size;
}
